// PoseSolver_6262062318060
// MI455X (gfx1250) — matrix operand order repaired and checked by reading, not compiled, not run on hardware
//
#include <hip/hip_runtime.h>
#include <hip/hip_bf16.h>

// ---------------------------------------------------------------------------
// PoseSolver for MI455X (gfx1250).  Pipeline:
//   k_pack      : f32 (B,C,N) -> f16 (B,N,C) LDS-tiled transpose (src & tgt emb)
//   k_norms     : xx[b,n] = sum_c emb^2  (coalesced over n)
//   k_gemm_wmma : pd[b,n,m] = 2*dot(f16 WMMA, f32 acc) - xx - yy
//                 LDS double-buffered B tile + register-prefetched A fragments,
//                 all 4 B fragments ds-loaded up front per K-step
//   k_rowstats  : rowmax / rowsumexp per row            (row softmax stats)
//   k_colstats  : colmax / colsumexp / col_sum(ssc)     (col softmax stats)
//   k_rowsum2   : row_sum of col-softmax
//   k_select    : 614-th smallest per batch (rank-by-count in LDS)
//   k_masks     : mask_src / mask_tgt (0/1 floats)
//   k_final     : sparse renorm + weighted gather of tgt -> src_corr, vsi
//   k_wnorm     : src_weight = vsi / sum(vsi)
//   k_svd       : weighted Kabsch (Jacobi 3x3) -> rot, trans
// ---------------------------------------------------------------------------

typedef __attribute__((ext_vector_type(16))) _Float16 v16h;
typedef __attribute__((ext_vector_type(8)))  _Float16 v8h;
typedef __attribute__((ext_vector_type(8)))  float    v8f;

#define BDIM 256

static constexpr int Bb   = 8;
static constexpr int Cc   = 512;
static constexpr int Nn   = 2048;
static constexpr int Mm   = 2048;
static constexpr int KSEL = 613;   // int(2048*0.3)=614 -> 0-indexed rank 613

// ---------------- block reduction helpers (blockDim.x == 256) ---------------
__device__ __forceinline__ float blk_sum(float v, float* red) {
    const int t = threadIdx.x;
    red[t] = v; __syncthreads();
    #pragma unroll
    for (int s = BDIM / 2; s > 0; s >>= 1) {
        if (t < s) red[t] += red[t + s];
        __syncthreads();
    }
    float r = red[0]; __syncthreads();
    return r;
}
__device__ __forceinline__ float blk_max(float v, float* red) {
    const int t = threadIdx.x;
    red[t] = v; __syncthreads();
    #pragma unroll
    for (int s = BDIM / 2; s > 0; s >>= 1) {
        if (t < s) red[t] = fmaxf(red[t], red[t + s]);
        __syncthreads();
    }
    float r = red[0]; __syncthreads();
    return r;
}

// ---------------- pack: (B,C,N) f32 -> (B,N,C) f16, LDS tile transpose ------
__global__ __launch_bounds__(256) void k_pack(const float* __restrict__ emb,
                                              _Float16* __restrict__ out) {
    __shared__ float tile[32][33];
    const int b  = blockIdx.z;
    const int n0 = blockIdx.x * 32, c0 = blockIdx.y * 32;
    const int tx = threadIdx.x,     ty = threadIdx.y;
    const float* s = emb + (size_t)b * Cc * Nn;
    #pragma unroll
    for (int i = 0; i < 4; ++i)
        tile[ty + 8 * i][tx] = s[(size_t)(c0 + ty + 8 * i) * Nn + n0 + tx];
    __syncthreads();
    _Float16* d = out + (size_t)b * Nn * Cc;
    #pragma unroll
    for (int i = 0; i < 4; ++i)
        d[(size_t)(n0 + ty + 8 * i) * Cc + c0 + tx] = (_Float16)tile[tx][ty + 8 * i];
}

// ---------------- squared norms over channel dim ----------------------------
__global__ __launch_bounds__(256) void k_norms(const float* __restrict__ emb,
                                               float* __restrict__ nrm) {
    const int b = blockIdx.y;
    const int n = blockIdx.x * BDIM + threadIdx.x;
    const float* p = emb + (size_t)b * Cc * Nn + n;
    float s = 0.f;
    for (int c = 0; c < Cc; ++c) { float v = p[(size_t)c * Nn]; s += v * v; }
    nrm[(size_t)b * Nn + n] = s;
}

// ---------------- WMMA GEMM: pd = 2*A.B - xx - yy ---------------------------
// Block = 8 waves -> 128(N) x 64(M) tile.  The 64x32 B tile is shared by all
// waves: staged in LDS (double-buffered, 2x4KB), one global b128 per thread
// per K-step, issued a full K-step ahead of its consuming WMMA.  A fragments
// (per-wave unique rows) are register-prefetched one K-step ahead.  All four
// B fragments are ds-loaded into distinct registers before the WMMA chain so
// the LDS latency drains with descending dscnt waits instead of serializing.
__global__ __launch_bounds__(256) void k_gemm_wmma(const _Float16* __restrict__ At,
                                                   const _Float16* __restrict__ Bt,
                                                   const float* __restrict__ xx,
                                                   const float* __restrict__ yy,
                                                   float* __restrict__ pd) {
    __shared__ _Float16 Bs[2][64 * 32];          // [buf][col*32 + k]

    const int tid  = threadIdx.x;
    const int lane = tid & 31;
    const int wave = tid >> 5;
    const int b    = blockIdx.z;
    const int rowBase = blockIdx.y * 128 + wave * 16;
    const int colBase = blockIdx.x * 64;
    const int l15  = lane & 15;
    const bool hi  = lane >= 16;

    // cooperative B staging: thread -> (col = tid>>2, kchunk = (tid&3)*8)
    const int stCol = tid >> 2;
    const int stK   = (tid & 3) * 8;
    const _Float16* stSrc = Bt + ((size_t)b * Mm + colBase + stCol) * Cc + stK;
    _Float16* stDst0 = &Bs[0][stCol * 32 + stK];
    _Float16* stDst1 = &Bs[1][stCol * 32 + stK];

    // A fragment pointer: lane holds row (rowBase+l15), K chunks
    // {0..15} (lo) / {16..31} (hi), matching the B fragment order.
    const _Float16* arow = At + ((size_t)b * Nn + rowBase + l15) * Cc + (hi ? 16 : 0);
    // B fragment read offset in LDS: col f*16+l15, K {0..15} (lo) / {16..31} (hi)
    const int bOff = hi ? 16 : 0;

    // ---- prologue: stage K-step 0, load A frag 0 ----
    v8h bStage = *(const v8h*)(stSrc);           // global -> regs
    v8h a0 = *(const v8h*)(arow);
    v8h a1 = *(const v8h*)(arow + 8);
    *(v8h*)stDst0 = bStage;                      // regs -> LDS buf 0
    __syncthreads();

    v8f acc[4] = {};
    #pragma unroll 4
    for (int kk = 0; kk < Cc; kk += 32) {
        const int cur = (kk >> 5) & 1;
        const bool last = (kk + 32 >= Cc);
        const int kn = last ? kk : kk + 32;      // clamped (redundant on last)

        // issue next K-step global loads early (latency hidden by WMMAs)
        v8h bNext = *(const v8h*)(stSrc + kn);
        v8h a0n = *(const v8h*)(arow + kn);
        v8h a1n = *(const v8h*)(arow + kn + 8);

        // assemble A fragment for this K-step
        v16h a;
        #pragma unroll
        for (int i = 0; i < 8; ++i) { a[i] = a0[i]; a[i + 8] = a1[i]; }

        // ds-load all four B fragments up front (distinct registers) ...
        const v16h bf0 = *(const v16h*)&Bs[cur][( 0 + l15) * 32 + bOff];
        const v16h bf1 = *(const v16h*)&Bs[cur][(16 + l15) * 32 + bOff];
        const v16h bf2 = *(const v16h*)&Bs[cur][(32 + l15) * 32 + bOff];
        const v16h bf3 = *(const v16h*)&Bs[cur][(48 + l15) * 32 + bOff];

        // ... then drain them through 4 independent accumulator chains
        acc[0] = __builtin_amdgcn_wmma_f32_16x16x32_f16(
            false, a, false, bf0, (short)0, acc[0], false, false);
        acc[1] = __builtin_amdgcn_wmma_f32_16x16x32_f16(
            false, a, false, bf1, (short)0, acc[1], false, false);
        acc[2] = __builtin_amdgcn_wmma_f32_16x16x32_f16(
            false, a, false, bf2, (short)0, acc[2], false, false);
        acc[3] = __builtin_amdgcn_wmma_f32_16x16x32_f16(
            false, a, false, bf3, (short)0, acc[3], false, false);

        // stage next K-step into the other LDS buffer
        if (!last) {
            __syncthreads();                     // all waves done with buf cur^1
            *(v8h*)(cur ? stDst0 : stDst1) = bNext;
            __syncthreads();                     // buf cur^1 ready
        }
        a0 = a0n; a1 = a1n;
    }

    // ---- epilogue: pd = 2*dot - xx - yy ----
    float xv[8];
    #pragma unroll
    for (int r = 0; r < 8; ++r)
        xv[r] = xx[b * Nn + rowBase + r + (hi ? 8 : 0)];
    #pragma unroll
    for (int f = 0; f < 4; ++f) {
        const int col = colBase + f * 16 + l15;
        const float yv = yy[b * Mm + col];
        #pragma unroll
        for (int r = 0; r < 8; ++r) {
            const int row = rowBase + r + (hi ? 8 : 0);
            pd[((size_t)b * Nn + row) * Mm + col] = 2.0f * acc[f][r] - xv[r] - yv;
        }
    }
}

// ---------------- row softmax stats -----------------------------------------
__global__ __launch_bounds__(256) void k_rowstats(const float* __restrict__ pd,
                                                  float* __restrict__ rmax,
                                                  float* __restrict__ rsum) {
    __shared__ float red[BDIM];
    const int b = blockIdx.y, n = blockIdx.x, t = threadIdx.x;
    const float* row = pd + ((size_t)b * Nn + n) * Mm;
    float mx = -3.4e38f;
    for (int m = t; m < Mm; m += BDIM) mx = fmaxf(mx, row[m]);
    mx = blk_max(mx, red);
    float s = 0.f;
    for (int m = t; m < Mm; m += BDIM) s += __expf(row[m] - mx);
    s = blk_sum(s, red);
    if (t == 0) { rmax[b * Nn + n] = mx; rsum[b * Nn + n] = s; }
}

// ---------------- col softmax stats + col_sum of row-softmax ----------------
__global__ __launch_bounds__(256) void k_colstats(const float* __restrict__ pd,
                                                  const float* __restrict__ rmax,
                                                  const float* __restrict__ rsum,
                                                  float* __restrict__ cmax,
                                                  float* __restrict__ csum,
                                                  float* __restrict__ colsum) {
    const int b = blockIdx.y;
    const int m = blockIdx.x * BDIM + threadIdx.x;
    const float* base = pd + (size_t)b * Nn * Mm + m;
    float mx = -3.4e38f;
    for (int n = 0; n < Nn; ++n) mx = fmaxf(mx, base[(size_t)n * Mm]);
    float se = 0.f, cs = 0.f;
    for (int n = 0; n < Nn; ++n) {
        const float v = base[(size_t)n * Mm];
        se += __expf(v - mx);
        cs += __expf(v - rmax[b * Nn + n]) / rsum[b * Nn + n];
    }
    cmax[(size_t)b * Mm + m] = mx;
    csum[(size_t)b * Mm + m] = se;
    colsum[(size_t)b * Mm + m] = cs;
}

// ---------------- row_sum of col-softmax ------------------------------------
__global__ __launch_bounds__(256) void k_rowsum2(const float* __restrict__ pd,
                                                 const float* __restrict__ cmax,
                                                 const float* __restrict__ csum,
                                                 float* __restrict__ rowsum2) {
    __shared__ float red[BDIM];
    const int b = blockIdx.y, n = blockIdx.x, t = threadIdx.x;
    const float* row = pd + ((size_t)b * Nn + n) * Mm;
    float s = 0.f;
    for (int m = t; m < Mm; m += BDIM)
        s += __expf(row[m] - cmax[(size_t)b * Mm + m]) / csum[(size_t)b * Mm + m];
    s = blk_sum(s, red);
    if (t == 0) rowsum2[b * Nn + n] = s;
}

// ---------------- k-th smallest per batch via rank counting -----------------
__global__ __launch_bounds__(256) void k_select(const float* __restrict__ colsum,
                                                const float* __restrict__ rowsum,
                                                float* __restrict__ kthc,
                                                float* __restrict__ kthr) {
    __shared__ float vals[2048];
    const int b = blockIdx.y;
    const bool doCol = (blockIdx.x == 0);
    const float* v = doCol ? (colsum + (size_t)b * Mm) : (rowsum + (size_t)b * Nn);
    const int len = doCol ? Mm : Nn;
    for (int i = threadIdx.x; i < len; i += BDIM) vals[i] = v[i];
    __syncthreads();
    for (int i = threadIdx.x; i < len; i += BDIM) {
        const float x = vals[i];
        int less = 0, leq = 0;
        for (int j = 0; j < len; ++j) {
            const float a = vals[j];
            less += (a < x);
            leq  += (a <= x);
        }
        if (less <= KSEL && KSEL < leq) {
            if (doCol) kthc[b] = x; else kthr[b] = x;
        }
    }
}

// ---------------- masks (0/1 floats, also exported) -------------------------
__global__ __launch_bounds__(256) void k_masks(const float* __restrict__ colsum,
                                               const float* __restrict__ rowsum,
                                               const float* __restrict__ kthc,
                                               const float* __restrict__ kthr,
                                               float* __restrict__ maskT,
                                               float* __restrict__ maskS,
                                               float* __restrict__ o_mt,
                                               float* __restrict__ o_ms) {
    const int b = blockIdx.y;
    const int i = blockIdx.x * BDIM + threadIdx.x;
    if (blockIdx.z == 0) {
        const float f = (colsum[(size_t)b * Mm + i] < kthc[b]) ? 1.0f : 0.0f;
        maskT[(size_t)b * Mm + i] = f;
        o_mt[(size_t)b * Mm + i]  = f;
    } else {
        const float f = (rowsum[(size_t)b * Nn + i] < kthr[b]) ? 1.0f : 0.0f;
        maskS[(size_t)b * Nn + i] = f;
        o_ms[(size_t)b * Nn + i]  = f;
    }
}

// ---------------- sparse renorm + weighted gather of tgt --------------------
__global__ __launch_bounds__(256) void k_final(const float* __restrict__ pd,
                                               const float* __restrict__ rmax,
                                               const float* __restrict__ rsum,
                                               const float* __restrict__ maskS,
                                               const float* __restrict__ maskT,
                                               const float* __restrict__ tgt,
                                               float* __restrict__ o_corr,
                                               float* __restrict__ vsi) {
    __shared__ float rowbuf[Mm];     // 8 KB pd row cached in LDS
    __shared__ float red[BDIM];
    const int b = blockIdx.y, n = blockIdx.x, t = threadIdx.x;
    const float* row = pd + ((size_t)b * Nn + n) * Mm;
    for (int m = t; m < Mm; m += BDIM) rowbuf[m] = row[m];
    __syncthreads();

    const float rm   = rmax[b * Nn + n];
    const float irs  = 1.0f / rsum[b * Nn + n];
    const float topv = irs;                     // K_TOP=1: max ssc in row = 1/rowsumexp
    const bool  ms   = maskS[b * Nn + n] > 0.5f;
    const float* mT  = maskT + (size_t)b * Mm;

    float loc = 0.f;
    for (int m = t; m < Mm; m += BDIM) {
        const float ssc = __expf(rowbuf[m] - rm) * irs;
        const float mk  = (ms || (mT[m] > 0.5f)) ? 1.0f : 0.0f;
        if (!((mk + ssc) < topv)) loc += ssc;
    }
    float colsp = blk_sum(loc, red);
    if (colsp < 1e-5f) colsp = 1e-5f;
    const float icol = 1.0f / colsp;

    const float* tg = tgt + (size_t)b * 3 * Mm;
    float vs = 0.f, c0 = 0.f, c1 = 0.f, c2 = 0.f;
    for (int m = t; m < Mm; m += BDIM) {
        const float ssc = __expf(rowbuf[m] - rm) * irs;
        const float mk  = (ms || (mT[m] > 0.5f)) ? 1.0f : 0.0f;
        const float w   = ((mk + ssc) < topv) ? 0.0f : ssc * icol;
        vs += w;
        c0 += tg[m] * w;
        c1 += tg[Mm + m] * w;
        c2 += tg[2 * Mm + m] * w;
    }
    vs = blk_sum(vs, red);
    c0 = blk_sum(c0, red);
    c1 = blk_sum(c1, red);
    c2 = blk_sum(c2, red);
    if (t == 0) {
        o_corr[((size_t)b * 3 + 0) * Nn + n] = c0;
        o_corr[((size_t)b * 3 + 1) * Nn + n] = c1;
        o_corr[((size_t)b * 3 + 2) * Nn + n] = c2;
        vsi[b * Nn + n] = ms ? 0.0f : vs;
    }
}

// ---------------- src_weight normalization ----------------------------------
__global__ __launch_bounds__(256) void k_wnorm(const float* __restrict__ vsi,
                                               float* __restrict__ o_w) {
    __shared__ float red[BDIM];
    const int b = blockIdx.x, t = threadIdx.x;
    float l = 0.f;
    for (int n = t; n < Nn; n += BDIM) l += vsi[(size_t)b * Nn + n];
    const float tot = blk_sum(l, red);
    const float inv = 1.0f / tot;
    for (int n = t; n < Nn; n += BDIM)
        o_w[(size_t)b * Nn + n] = vsi[(size_t)b * Nn + n] * inv;
}

// ---------------- weighted Kabsch / 3x3 SVD ---------------------------------
__global__ __launch_bounds__(256) void k_svd(const float* __restrict__ src,
                                             const float* __restrict__ corr,
                                             const float* __restrict__ wgt,
                                             float* __restrict__ rot_out,
                                             float* __restrict__ trans_out) {
    __shared__ float red[BDIM];
    const int b = blockIdx.x, t = threadIdx.x;
    const float* sa = src  + (size_t)b * 3 * Nn;   // a[n][d] = sa[d*Nn+n]
    const float* sb = corr + (size_t)b * 3 * Nn;
    const float* w  = wgt  + (size_t)b * Nn;

    float lw = 0.f;
    for (int n = t; n < Nn; n += BDIM) lw += w[n];
    const float wsum = blk_sum(lw, red);
    const float winv = 1.0f / (wsum + 1e-5f);

    float ca[3], cb[3];
    for (int d = 0; d < 3; ++d) {
        float la = 0.f, lb = 0.f;
        for (int n = t; n < Nn; n += BDIM) {
            const float wn = w[n] * winv;
            la += sa[d * Nn + n] * wn;
            lb += sb[d * Nn + n] * wn;
        }
        ca[d] = blk_sum(la, red);
        cb[d] = blk_sum(lb, red);
    }
    float cov[3][3];
    for (int d = 0; d < 3; ++d)
        for (int e = 0; e < 3; ++e) {
            float lc = 0.f;
            for (int n = t; n < Nn; n += BDIM)
                lc += (sa[d * Nn + n] - ca[d]) * (sb[e * Nn + n] - cb[e]) * (w[n] * winv);
            cov[d][e] = blk_sum(lc, red);
        }

    if (t == 0) {
        // Jacobi eigensolve of S = cov^T cov  ->  V, sigma^2
        float S[3][3], V[3][3];
        for (int i = 0; i < 3; ++i)
            for (int j = 0; j < 3; ++j) {
                float s = 0.f;
                for (int k = 0; k < 3; ++k) s += cov[k][i] * cov[k][j];
                S[i][j] = s;
                V[i][j] = (i == j) ? 1.f : 0.f;
            }
        for (int sweep = 0; sweep < 10; ++sweep)
            for (int pi = 0; pi < 3; ++pi) {
                const int p = (pi == 2) ? 1 : 0;
                const int q = (pi == 0) ? 1 : 2;
                const float apq = S[p][q];
                if (fabsf(apq) < 1e-20f) continue;
                const float theta = 0.5f * (S[q][q] - S[p][p]) / apq;
                const float tt = copysignf(1.0f, theta) /
                                 (fabsf(theta) + sqrtf(1.0f + theta * theta));
                const float c = rsqrtf(1.0f + tt * tt);
                const float s = tt * c;
                for (int k = 0; k < 3; ++k) {
                    const float skp = S[k][p], skq = S[k][q];
                    S[k][p] = c * skp - s * skq;
                    S[k][q] = s * skp + c * skq;
                }
                for (int k = 0; k < 3; ++k) {
                    const float spk = S[p][k], sqk = S[q][k];
                    S[p][k] = c * spk - s * sqk;
                    S[q][k] = s * spk + c * sqk;
                }
                for (int k = 0; k < 3; ++k) {
                    const float vkp = V[k][p], vkq = V[k][q];
                    V[k][p] = c * vkp - s * vkq;
                    V[k][q] = s * vkp + c * vkq;
                }
            }
        float lam[3] = { S[0][0], S[1][1], S[2][2] };
        int idx[3] = { 0, 1, 2 };
        for (int i = 0; i < 2; ++i)
            for (int j = i + 1; j < 3; ++j)
                if (lam[idx[j]] > lam[idx[i]]) { int tmp = idx[i]; idx[i] = idx[j]; idx[j] = tmp; }
        float Vm[3][3], Um[3][3];
        for (int i = 0; i < 3; ++i) {
            const int cidx = idx[i];
            const float sv = sqrtf(fmaxf(lam[cidx], 0.f));
            const float isv = (sv > 1e-12f) ? 1.0f / sv : 0.0f;
            for (int k = 0; k < 3; ++k) Vm[k][i] = V[k][cidx];
            for (int r = 0; r < 3; ++r) {
                float s = 0.f;
                for (int k = 0; k < 3; ++k) s += cov[r][k] * Vm[k][i];
                Um[r][i] = s * isv;
            }
        }
        float R[3][3];
        for (int i = 0; i < 3; ++i)
            for (int j = 0; j < 3; ++j) {
                float s = 0.f;
                for (int k = 0; k < 3; ++k) s += Vm[i][k] * Um[j][k];
                R[i][j] = s;
            }
        const float det = R[0][0] * (R[1][1] * R[2][2] - R[1][2] * R[2][1])
                        - R[0][1] * (R[1][0] * R[2][2] - R[1][2] * R[2][0])
                        + R[0][2] * (R[1][0] * R[2][1] - R[1][1] * R[2][0]);
        if (det <= 0.f) {
            for (int i = 0; i < 3; ++i) Vm[i][2] = -Vm[i][2];
            for (int i = 0; i < 3; ++i)
                for (int j = 0; j < 3; ++j) {
                    float s = 0.f;
                    for (int k = 0; k < 3; ++k) s += Vm[i][k] * Um[j][k];
                    R[i][j] = s;
                }
        }
        for (int i = 0; i < 3; ++i)
            for (int j = 0; j < 3; ++j)
                rot_out[b * 9 + i * 3 + j] = R[i][j];
        for (int i = 0; i < 3; ++i) {
            float s = 0.f;
            for (int j = 0; j < 3; ++j) s += R[i][j] * ca[j];
            trans_out[b * 3 + i] = -s + cb[i];
        }
    }
}

// ---------------------------------------------------------------------------
extern "C" void kernel_launch(void* const* d_in, const int* in_sizes, int n_in,
                              void* d_out, int out_size, void* d_ws, size_t ws_size,
                              hipStream_t stream) {
    (void)in_sizes; (void)n_in; (void)out_size; (void)ws_size;
    const float* src  = (const float*)d_in[0];   // (B,3,N)
    const float* tgt  = (const float*)d_in[1];   // (B,3,M)
    const float* semb = (const float*)d_in[2];   // (B,C,N)
    const float* temb = (const float*)d_in[3];   // (B,C,M)
    float* out = (float*)d_out;

    // output regions (flat concat in return order)
    float* o_corr = out;                               // B*3*N
    float* o_w    = o_corr + (size_t)Bb * 3 * Nn;      // B*N
    float* o_ms   = o_w    + (size_t)Bb * Nn;          // B*N
    float* o_mt   = o_ms   + (size_t)Bb * Nn;          // B*M
    float* o_rot  = o_mt   + (size_t)Bb * Mm;          // B*9
    float* o_tr   = o_rot  + (size_t)Bb * 9;           // B*3

    // workspace carve-up (~168 MB)
    char* ws = (char*)d_ws;
    size_t off = 0;
    auto carve = [&](size_t bytes) {
        void* p = ws + off;
        off += (bytes + 255) & ~(size_t)255;
        return p;
    };
    _Float16* srcT = (_Float16*)carve((size_t)Bb * Nn * Cc * sizeof(_Float16));
    _Float16* tgtT = (_Float16*)carve((size_t)Bb * Mm * Cc * sizeof(_Float16));
    float* pd      = (float*)carve((size_t)Bb * Nn * Mm * sizeof(float));
    float* xx      = (float*)carve((size_t)Bb * Nn * sizeof(float));
    float* yy      = (float*)carve((size_t)Bb * Mm * sizeof(float));
    float* rmax    = (float*)carve((size_t)Bb * Nn * sizeof(float));
    float* rsum    = (float*)carve((size_t)Bb * Nn * sizeof(float));
    float* cmax    = (float*)carve((size_t)Bb * Mm * sizeof(float));
    float* csum    = (float*)carve((size_t)Bb * Mm * sizeof(float));
    float* colsum  = (float*)carve((size_t)Bb * Mm * sizeof(float));
    float* rowsum2 = (float*)carve((size_t)Bb * Nn * sizeof(float));
    float* kthc    = (float*)carve((size_t)Bb * sizeof(float));
    float* kthr    = (float*)carve((size_t)Bb * sizeof(float));
    float* maskT   = (float*)carve((size_t)Bb * Mm * sizeof(float));
    float* maskS   = (float*)carve((size_t)Bb * Nn * sizeof(float));
    float* vsi     = (float*)carve((size_t)Bb * Nn * sizeof(float));

    k_pack<<<dim3(Nn / 32, Cc / 32, Bb), dim3(32, 8), 0, stream>>>(semb, srcT);
    k_pack<<<dim3(Mm / 32, Cc / 32, Bb), dim3(32, 8), 0, stream>>>(temb, tgtT);
    k_norms<<<dim3(Nn / BDIM, Bb), BDIM, 0, stream>>>(semb, xx);
    k_norms<<<dim3(Mm / BDIM, Bb), BDIM, 0, stream>>>(temb, yy);
    k_gemm_wmma<<<dim3(Mm / 64, Nn / 128, Bb), BDIM, 0, stream>>>(srcT, tgtT, xx, yy, pd);
    k_rowstats<<<dim3(Nn, Bb), BDIM, 0, stream>>>(pd, rmax, rsum);
    k_colstats<<<dim3(Mm / BDIM, Bb), BDIM, 0, stream>>>(pd, rmax, rsum, cmax, csum, colsum);
    k_rowsum2<<<dim3(Nn, Bb), BDIM, 0, stream>>>(pd, cmax, csum, rowsum2);
    k_select<<<dim3(2, Bb), BDIM, 0, stream>>>(colsum, rowsum2, kthc, kthr);
    k_masks<<<dim3(Mm / BDIM, Bb, 2), BDIM, 0, stream>>>(colsum, rowsum2, kthc, kthr,
                                                         maskT, maskS, o_mt, o_ms);
    k_final<<<dim3(Nn, Bb), BDIM, 0, stream>>>(pd, rmax, rsum, maskS, maskT, tgt, o_corr, vsi);
    k_wnorm<<<dim3(Bb), BDIM, 0, stream>>>(vsi, o_w);
    k_svd<<<dim3(Bb), BDIM, 0, stream>>>(src, o_corr, o_w, o_rot, o_tr);
}
